// AttentionHead_22582938042791
// MI455X (gfx1250) — matrix operand order repaired and checked by reading, not compiled, not run on hardware
//
#include <hip/hip_runtime.h>

typedef __bf16 bf16;
typedef __attribute__((ext_vector_type(16))) __bf16 v16bf;
typedef __attribute__((ext_vector_type(8)))  __bf16 v8bf;
typedef __attribute__((ext_vector_type(8)))  float  v8f;
typedef __attribute__((ext_vector_type(4)))  float  v4f;

#define S_LEN 2048
#define E_DIM 1024
#define H_DIM 64
#define B_N   4

#define KSPLIT 2          // waves per workgroup, each owns S_LEN/KSPLIT keys
#define PROW   72         // ptile row stride (bf16) : 64 cols + pad

// ---------------------------------------------------------------------------
// Stage 0: repack fp32 weights [E][H] -> bf16 transposed [H][E] so that
// WMMA B-fragments (contiguous k order used here: lane = column n, lanes 16-31 carry
// K=16..31) are a single contiguous 32-byte load per lane.
// ---------------------------------------------------------------------------
__global__ void wconv_kernel(const float* __restrict__ Wq,
                             const float* __restrict__ Wk,
                             const float* __restrict__ Wv,
                             bf16* __restrict__ wT) {
  int tid = blockIdx.x * blockDim.x + threadIdx.x;  // 3 * 64 * 1024 threads
  int z = tid >> 16;
  int r = tid & 65535;
  int h = r >> 10;
  int e = r & 1023;
  const float* W = (z == 0) ? Wq : ((z == 1) ? Wk : Wv);
  wT[(size_t)z * H_DIM * E_DIM + (size_t)h * E_DIM + e] = (bf16)W[e * H_DIM + h];
}

// ---------------------------------------------------------------------------
// Stage 1: projections  x[B*S,E] @ W[E,H] + b  -> bf16
//   z=0: Q -> qp[B*S][H]   z=1: K -> kp[B*S][H]   z=2: V -> vpT[B][H][S]
// ---------------------------------------------------------------------------
__global__ void proj_kernel(const float* __restrict__ xq,
                            const float* __restrict__ xk,
                            const float* __restrict__ xv,
                            const bf16* __restrict__ wT,
                            const float* __restrict__ bq,
                            const float* __restrict__ bk,
                            const float* __restrict__ bv,
                            bf16* __restrict__ qp,
                            bf16* __restrict__ kp,
                            bf16* __restrict__ vpT) {
  const int z = blockIdx.z;
  const float* x    = (z == 0) ? xq : ((z == 1) ? xk : xv);
  const float* bias = (z == 0) ? bq : ((z == 1) ? bk : bv);
  const bf16*  w    = wT + (size_t)z * H_DIM * E_DIM;

  const int lane = threadIdx.x;     // 0..31 (wave32)
  const int half = lane >> 4;
  const int l16  = lane & 15;
  const int q0   = blockIdx.x * 16; // row tile over flattened B*S

  v8f acc[4] = {};
  const float* arow = x + (size_t)(q0 + l16) * E_DIM;

  for (int k0 = 0; k0 < E_DIM; k0 += 32) {
    v4f r0 = *(const v4f*)(arow + k0 + half * 16);
    v4f r1 = *(const v4f*)(arow + k0 + half * 16 + 4);
    v4f r2 = *(const v4f*)(arow + k0 + half * 16 + 8);
    v4f r3 = *(const v4f*)(arow + k0 + half * 16 + 12);
    v16bf a;
#pragma unroll
    for (int i = 0; i < 4; i++) {
      a[i]      = (bf16)r0[i];
      a[4 + i]  = (bf16)r1[i];
      a[8 + i]  = (bf16)r2[i];
      a[12 + i] = (bf16)r3[i];
    }
#pragma unroll
    for (int nt = 0; nt < 4; nt++) {
      v16bf bfr = *(const v16bf*)(w + (size_t)(nt * 16 + l16) * E_DIM + k0 + half * 16);
      acc[nt] = __builtin_amdgcn_wmma_f32_16x16x32_bf16(
          false, a, false, bfr, (short)0, acc[nt], false, false);
    }
  }

#pragma unroll
  for (int nt = 0; nt < 4; nt++) {
    float bval = bias[nt * 16 + l16];
#pragma unroll
    for (int r = 0; r < 8; r++) {
      float val = acc[nt][r] + bval;
      int m  = r + 8 * half;
      int gs = q0 + m;
      int h  = nt * 16 + l16;
      if (z == 2) {
        int bb = gs >> 11, s = gs & (S_LEN - 1);
        vpT[((size_t)bb * H_DIM + h) * S_LEN + s] = (bf16)val;
      } else if (z == 0) {
        qp[(size_t)gs * H_DIM + h] = (bf16)val;
      } else {
        kp[(size_t)gs * H_DIM + h] = (bf16)val;
      }
    }
  }
}

// ---------------------------------------------------------------------------
// Stage 2: flash attention.
// Workgroup = 2 waves; each wave owns one (batch, 16-query tile) and HALF the
// keys (1024), running a private online softmax over 64-key tiles:
//   - 8 WMMAs  -> four 16x16 score tiles (K=64 in two 32-chunks)
//   - one shuffle-reduction pass + one O rescale per 64 keys (amortized 2x
//     vs 32-key tiles; softmax VALU was the inner-loop bottleneck)
//   - P (16x64 bf16) through LDS to convert C-layout -> A-layout
//   - 8 WMMAs  -> O += P @ V (V pre-transposed [B][H][S] bf16)
// Final flash combine of the two key-halves through LDS.
// ---------------------------------------------------------------------------
__global__ void attn_kernel(const bf16* __restrict__ qp,
                            const bf16* __restrict__ kp,
                            const bf16* __restrict__ vpT,
                            float* __restrict__ out) {
  __shared__ __align__(16) bf16 ptile[KSPLIT * 16 * PROW];
  __shared__ float cmax[KSPLIT][16];
  __shared__ float csum[KSPLIT][16];
  __shared__ float cO[KSPLIT][16][H_DIM];

  const int w    = threadIdx.x >> 5;   // wave id: which key half
  const int lane = threadIdx.x & 31;
  const int half = lane >> 4;
  const int l16  = lane & 15;
  const int b    = blockIdx.y;
  const int q0   = blockIdx.x * 16;

  const bf16* qb = qp  + (size_t)b * S_LEN * H_DIM;
  const bf16* kb = kp  + (size_t)b * S_LEN * H_DIM;
  const bf16* vb = vpT + (size_t)b * H_DIM * S_LEN;
  bf16* pt = ptile + w * 16 * PROW;

  // Q A-fragments for h0 = 0 and 32 (loaded once)
  v16bf qf[2];
#pragma unroll
  for (int hh = 0; hh < 2; hh++) {
    const bf16* qr = qb + (size_t)(q0 + l16) * H_DIM + hh * 32;
    v8bf lo = *(const v8bf*)(qr + half * 16);
    v8bf hi = *(const v8bf*)(qr + half * 16 + 8);
#pragma unroll
    for (int i = 0; i < 8; i++) { qf[hh][i] = lo[i]; qf[hh][8 + i] = hi[i]; }
  }

  v8f O[4] = {};
  float rmax[8], rsum[8];
#pragma unroll
  for (int r = 0; r < 8; r++) { rmax[r] = -3.0e38f; rsum[r] = 0.0f; }

  const int kbeg = w * (S_LEN / KSPLIT);
  const int kend = kbeg + (S_LEN / KSPLIT);

  for (int k0 = kbeg; k0 < kend; k0 += 64) {
    // ---- scores: four 16x16 tiles over 64 keys ----
    v8f Sx[4] = {};
#pragma unroll
    for (int hh = 0; hh < 2; hh++) {
#pragma unroll
      for (int nt = 0; nt < 4; nt++) {
        v16bf kf = *(const v16bf*)(kb + (size_t)(k0 + nt * 16 + l16) * H_DIM +
                                   hh * 32 + half * 16);
        Sx[nt] = __builtin_amdgcn_wmma_f32_16x16x32_bf16(
            false, qf[hh], false, kf, (short)0, Sx[nt], false, false);
      }
    }

    // ---- online softmax: one reduction pass per 64 keys ----
#pragma unroll
    for (int r = 0; r < 8; r++) {
      float tm = fmaxf(fmaxf(Sx[0][r], Sx[1][r]), fmaxf(Sx[2][r], Sx[3][r]));
#pragma unroll
      for (int off = 1; off < 16; off <<= 1) tm = fmaxf(tm, __shfl_xor(tm, off, 32));
      float nm = fmaxf(rmax[r], tm);
      float p0 = __expf(Sx[0][r] - nm);
      float p1 = __expf(Sx[1][r] - nm);
      float p2 = __expf(Sx[2][r] - nm);
      float p3 = __expf(Sx[3][r] - nm);
      float ts = (p0 + p1) + (p2 + p3);
#pragma unroll
      for (int off = 1; off < 16; off <<= 1) ts += __shfl_xor(ts, off, 32);
      float sc = __expf(rmax[r] - nm);
      rsum[r] = rsum[r] * sc + ts;
      rmax[r] = nm;
#pragma unroll
      for (int nt = 0; nt < 4; nt++) O[nt][r] *= sc;
      int m = r + 8 * half;
      pt[m * PROW +  0 + l16] = (bf16)p0;
      pt[m * PROW + 16 + l16] = (bf16)p1;
      pt[m * PROW + 32 + l16] = (bf16)p2;
      pt[m * PROW + 48 + l16] = (bf16)p3;
    }
    __syncthreads();

    // ---- O += P (16x64) @ V (64x64): two 32-key chunks ----
#pragma unroll
    for (int c = 0; c < 2; c++) {
      v16bf pf;
      const bf16* pr = pt + l16 * PROW + c * 32;
      v8bf lo = *(const v8bf*)(pr + half * 16);
      v8bf hi = *(const v8bf*)(pr + half * 16 + 8);
#pragma unroll
      for (int i = 0; i < 8; i++) { pf[i] = lo[i]; pf[8 + i] = hi[i]; }
#pragma unroll
      for (int nt = 0; nt < 4; nt++) {
        v16bf vf = *(const v16bf*)(vb + (size_t)(nt * 16 + l16) * S_LEN +
                                   k0 + c * 32 + half * 16);
        O[nt] = __builtin_amdgcn_wmma_f32_16x16x32_bf16(
            false, pf, false, vf, (short)0, O[nt], false, false);
      }
    }
    __syncthreads();
  }

  // ---- publish per-wave partials ----
#pragma unroll
  for (int r = 0; r < 8; r++) {
    int m = r + 8 * half;
    if (l16 == 0) { cmax[w][m] = rmax[r]; csum[w][m] = rsum[r]; }
#pragma unroll
    for (int nt = 0; nt < 4; nt++) cO[w][m][nt * 16 + l16] = O[nt][r];
  }
  __syncthreads();

  // ---- flash combine of the two key-halves; wave 0 writes output ----
  if (w == 0) {
    for (int i = lane; i < 16 * H_DIM; i += 32) {
      int row = i >> 6, col = i & (H_DIM - 1);
      float m0 = cmax[0][row], m1 = cmax[1][row];
      float M  = fmaxf(m0, m1);
      float a0 = __expf(m0 - M), a1 = __expf(m1 - M);
      float denom = csum[0][row] * a0 + csum[1][row] * a1;
      float val = (cO[0][row][col] * a0 + cO[1][row][col] * a1) / denom;
      out[((size_t)b * S_LEN + q0 + row) * H_DIM + col] = val;
    }
  }
}

// ---------------------------------------------------------------------------
extern "C" void kernel_launch(void* const* d_in, const int* in_sizes, int n_in,
                              void* d_out, int out_size, void* d_ws, size_t ws_size,
                              hipStream_t stream) {
  const float* query = (const float*)d_in[0];
  const float* key   = (const float*)d_in[1];
  const float* value = (const float*)d_in[2];
  const float* Wq    = (const float*)d_in[3];
  const float* bq    = (const float*)d_in[4];
  const float* Wk    = (const float*)d_in[5];
  const float* bk    = (const float*)d_in[6];
  const float* Wv    = (const float*)d_in[7];
  const float* bv    = (const float*)d_in[8];
  float* out = (float*)d_out;

  char* ws = (char*)d_ws;
  bf16* qp  = (bf16*)(ws);                 // 1 MB  [B*S][H]
  bf16* kp  = (bf16*)(ws + (1u << 20));    // 1 MB  [B*S][H]
  bf16* vpT = (bf16*)(ws + (2u << 20));    // 1 MB  [B][H][S]
  bf16* wT  = (bf16*)(ws + (3u << 20));    // 384 KB  3 x [H][E]

  wconv_kernel<<<dim3((3 * H_DIM * E_DIM) / 256), dim3(256), 0, stream>>>(Wq, Wk, Wv, wT);
  proj_kernel<<<dim3((B_N * S_LEN) / 16, 1, 3), dim3(32), 0, stream>>>(
      query, key, value, wT, bq, bk, bv, qp, kp, vpT);
  attn_kernel<<<dim3(S_LEN / 16, B_N), dim3(32 * KSPLIT), 0, stream>>>(qp, kp, vpT, out);
}
